// GAT_Mixer_17497696764597
// MI455X (gfx1250) — matrix operand order repaired and checked by reading, not compiled, not run on hardware
//
#include <hip/hip_runtime.h>
#include <math.h>
#include <stdint.h>

// ---------------------------------------------------------------------------
// Types for CDNA5 WMMA
// ---------------------------------------------------------------------------
typedef __bf16 bf16_t;
typedef __attribute__((ext_vector_type(16))) __bf16 bf16x16;
typedef __attribute__((ext_vector_type(8)))  float  floatx8;

#define GN   4096   // nodes
#define TM   128    // block tile M
#define TN   128    // block tile N
#define TK   32     // k step
#define LDA  40     // padded A-tile stride (elements); 40*2B = 80B, multiple of 16B

// ---------------------------------------------------------------------------
// Dense GEMM: C[M x Nc] = A[M x K] @ W[K x Nc] (+ bias), bf16 WMMA, fp32 acc.
// Block tile 128x128, 8 waves, each wave a 32x64 tile (2x4 WMMA accumulators).
// M % 128 == 0, K % 32 == 0, Nc % 128 == 0 (true for all layers here).
// ---------------------------------------------------------------------------
__global__ __launch_bounds__(256)
void gemm_bf16_wmma(const float* __restrict__ A, const float* __restrict__ W,
                    float* __restrict__ C, const float* __restrict__ bias,
                    int M, int K, int Nc) {
    __shared__ __align__(16) bf16_t As[TM * LDA];   // row-major, K contiguous
    __shared__ __align__(16) bf16_t Bs[TK * TN];    // K-pair packed: dword p*TN+n = {k=2p, 2p+1} at col n

    const int t    = threadIdx.x;
    const int lane = t & 31;
    const int wave = t >> 5;
    const int lm   = lane & 15;      // N-col / M-row within 16
    const int lh   = lane >> 4;      // lane half selects K-group / M+8
    const int row0 = blockIdx.y * TM;
    const int col0 = blockIdx.x * TN;
    const int wr   = (wave & 3) * 32;   // wave row offset in block tile (4 row groups)
    const int wc   = (wave >> 2) * 64;  // wave col offset in block tile (2 col groups)

    floatx8 acc[2][4] = {};

    for (int kk = 0; kk < K; kk += TK) {
        // ---- stage A tile (TM x TK) fp32 -> bf16 ----
        #pragma unroll
        for (int i = 0; i < (TM * TK) / 256; ++i) {
            int e = t + i * 256;
            int r = e >> 5, c = e & 31;
            As[r * LDA + c] = (bf16_t)A[(size_t)(row0 + r) * K + kk + c];
        }
        // ---- stage B tile (TK x TN), packed in K-pairs for fragment layout ----
        #pragma unroll
        for (int i = 0; i < (TK * TN) / 256; ++i) {
            int e = t + i * 256;
            int kr = e >> 7, n = e & 127;
            bf16_t v = (bf16_t)W[(size_t)(kk + kr) * Nc + col0 + n];
            Bs[(kr >> 1) * (2 * TN) + n * 2 + (kr & 1)] = v;
        }
        __syncthreads();

        // ---- load A fragments (ISA 16-bit A 16x32 layout) ----
        bf16x16 afrag[2];
        #pragma unroll
        for (int i = 0; i < 2; ++i) {
            const bf16_t* ap = &As[(wr + i * 16 + lm) * LDA + lh * 8];
            union { bf16x16 v; uint4 q[2]; } ua;
            ua.q[0] = *(const uint4*)(ap);        // K = base .. base+7
            ua.q[1] = *(const uint4*)(ap + 16);   // K = base+16 .. base+23
            afrag[i] = ua.v;
        }
        // ---- load B fragments (VGPR g<4: K = 8*lh+2g,+1; g>=4: K = 16+8*lh+2(g-4),+1) ----
        const uint32_t* bp = (const uint32_t*)Bs;
        bf16x16 bfrag[4];
        #pragma unroll
        for (int j = 0; j < 4; ++j) {
            union { bf16x16 v; uint32_t d[8]; } ub;
            int n = wc + j * 16 + lm;
            #pragma unroll
            for (int g = 0; g < 8; ++g)
                ub.d[g] = bp[(lh * 4 + g + (g & 4)) * TN + n];
            bfrag[j] = ub.v;
        }
        // ---- 2x4 WMMA ----
        #pragma unroll
        for (int i = 0; i < 2; ++i)
            #pragma unroll
            for (int j = 0; j < 4; ++j)
                acc[i][j] = __builtin_amdgcn_wmma_f32_16x16x32_bf16(
                    false, afrag[i], false, bfrag[j],
                    (short)0, acc[i][j], false, false);
        __syncthreads();
    }

    // ---- epilogue: C/D layout (VGPR v -> M = v + lh*8, col = lm) ----
    #pragma unroll
    for (int i = 0; i < 2; ++i) {
        #pragma unroll
        for (int j = 0; j < 4; ++j) {
            int cn = col0 + wc + j * 16 + lm;
            float bv = bias ? bias[cn] : 0.0f;
            #pragma unroll
            for (int v = 0; v < 8; ++v) {
                int rm = row0 + wr + i * 16 + lh * 8 + v;
                C[(size_t)rm * Nc + cn] = acc[i][j][v] + bv;
            }
        }
    }
}

// ---------------------------------------------------------------------------
// Small helpers
// ---------------------------------------------------------------------------
__device__ inline float wave_sum(float v) {
    #pragma unroll
    for (int off = 16; off; off >>= 1) v += __shfl_xor(v, off, 32);
    return v;
}
__device__ inline float wave_max(float v) {
    #pragma unroll
    for (int off = 16; off; off >>= 1) v = fmaxf(v, __shfl_xor(v, off, 32));
    return v;
}

__global__ void fill_i32(int* p, long long n, int v) {
    long long i = (long long)blockIdx.x * blockDim.x + threadIdx.x;
    if (i < n) p[i] = v;
}
__global__ void fill_bias_k(float* out, const float* bias, long long n, int D) {
    long long i = (long long)blockIdx.x * blockDim.x + threadIdx.x;
    if (i < n) out[i] = bias[(int)(i % D)];
}
__global__ void elu_k(float* p, long long n) {
    long long i = (long long)blockIdx.x * blockDim.x + threadIdx.x;
    if (i < n) { float x = p[i]; p[i] = x > 0.f ? x : (expf(x) - 1.f); }
}

// ---------------------------------------------------------------------------
// CSR build (edges grouped by dst).  cnt also yields GCN degree (cnt+1 w/ loop).
// ---------------------------------------------------------------------------
__global__ void cnt_count_k(const int* __restrict__ dst, int E, int* cnt) {
    int i = blockIdx.x * blockDim.x + threadIdx.x;
    if (i < E) atomicAdd(&cnt[dst[i]], 1);
}
__global__ void make_dinv_k(const int* __restrict__ cnt, float* dinv, int n) {
    int i = blockIdx.x * blockDim.x + threadIdx.x;
    if (i < n) dinv[i] = rsqrtf((float)(cnt[i] + 1));   // +1 self-loop, always >= 1
}
// single-block exclusive scan of 4096 counts -> row_start[0..4096]
__global__ __launch_bounds__(256)
void scan4096_k(const int* __restrict__ cnt, int* __restrict__ row_start, int n) {
    __shared__ int part[256];
    int t = threadIdx.x;
    int base = t * 16;
    int local[16];
    int s = 0;
    #pragma unroll
    for (int i = 0; i < 16; ++i) { local[i] = s; s += cnt[base + i]; }
    part[t] = s;
    __syncthreads();
    #pragma unroll
    for (int off = 1; off < 256; off <<= 1) {
        int v = (t >= off) ? part[t - off] : 0;
        __syncthreads();
        part[t] += v;
        __syncthreads();
    }
    int prev = (t == 0) ? 0 : part[t - 1];
    #pragma unroll
    for (int i = 0; i < 16; ++i) row_start[base + i] = prev + local[i];
    if (t == 255) row_start[n] = part[255];
}
__global__ void csr_scatter_k(const int* __restrict__ dst, int E,
                              const int* __restrict__ row_start, int* cursor, int* eid) {
    int i = blockIdx.x * blockDim.x + threadIdx.x;
    if (i >= E) return;
    int d = dst[i];
    int p = row_start[d] + atomicAdd(&cursor[d], 1);
    eid[p] = i;
}

// ---------------------------------------------------------------------------
// GAT: per-node attention dots, then wave-per-node segment softmax (no atomics)
// ---------------------------------------------------------------------------
__global__ __launch_bounds__(256)
void gat_alpha_k(const float* __restrict__ H, const float* __restrict__ a_src,
                 const float* __restrict__ a_dst, int n, int D,
                 float* asrc, float* adst) {
    int w = (int)((blockIdx.x * blockDim.x + threadIdx.x) >> 5);
    int lane = threadIdx.x & 31;
    if (w >= n) return;
    const float* hp = H + (size_t)w * D;
    float s1 = 0.f, s2 = 0.f;
    for (int i = lane; i < D; i += 32) {
        float h = hp[i];
        s1 += h * a_src[i];
        s2 += h * a_dst[i];
    }
    s1 = wave_sum(s1); s2 = wave_sum(s2);
    if (lane == 0) { asrc[w] = s1; adst[w] = s2; }
}
__device__ inline float lrelu02(float x) { return x > 0.f ? x : 0.2f * x; }

__global__ __launch_bounds__(256)
void gat_softmax_k(const int* __restrict__ row_start, const int* __restrict__ eid,
                   const int* __restrict__ src,
                   const float* __restrict__ asrc, const float* __restrict__ adst,
                   int n, float* __restrict__ alpha, float* __restrict__ aself) {
    int w = (int)((blockIdx.x * blockDim.x + threadIdx.x) >> 5);
    int lane = threadIdx.x & 31;
    if (w >= n) return;
    int p0 = row_start[w], p1 = row_start[w + 1];
    float ad = adst[w];
    float lself = lrelu02(asrc[w] + ad);
    float m = lself;
    for (int p = p0 + lane; p < p1; p += 32)
        m = fmaxf(m, lrelu02(asrc[src[eid[p]]] + ad));
    m = wave_max(m);
    float ssum = (lane == 0) ? expf(lself - m) : 0.f;
    for (int p = p0 + lane; p < p1; p += 32)
        ssum += expf(lrelu02(asrc[src[eid[p]]] + ad) - m);
    ssum = wave_sum(ssum);
    float inv = 1.f / (ssum + 1e-16f);
    for (int p = p0 + lane; p < p1; p += 32) {
        int e = eid[p];
        alpha[e] = expf(lrelu02(asrc[src[e]] + ad) - m) * inv;
    }
    if (lane == 0) aself[w] = expf(lself - m) * inv;
}

// ---------------------------------------------------------------------------
// Transformer: wave-per-edge q.k logits, then wave-per-node softmax in place
// ---------------------------------------------------------------------------
__global__ __launch_bounds__(256)
void tr_logits_k(const int* __restrict__ src, const int* __restrict__ dst, int E,
                 const float* __restrict__ Q, const float* __restrict__ Kf,
                 int D, float scale, float* __restrict__ elog) {
    int w = (int)((blockIdx.x * blockDim.x + threadIdx.x) >> 5);
    int lane = threadIdx.x & 31;
    if (w >= E) return;
    int s = src[w], d = dst[w];
    const float* qp = Q + (size_t)d * D;
    const float* kp = Kf + (size_t)s * D;
    float acc = 0.f;
    for (int i = lane; i < D; i += 32) acc += qp[i] * kp[i];
    acc = wave_sum(acc);
    if (lane == 0) elog[w] = acc * scale;
}
__global__ __launch_bounds__(256)
void tr_softmax_k(const int* __restrict__ row_start, const int* __restrict__ eid,
                  int n, float* __restrict__ elog) {
    int w = (int)((blockIdx.x * blockDim.x + threadIdx.x) >> 5);
    int lane = threadIdx.x & 31;
    if (w >= n) return;
    int p0 = row_start[w], p1 = row_start[w + 1];
    float m = -3.4e38f;
    for (int p = p0 + lane; p < p1; p += 32) m = fmaxf(m, elog[eid[p]]);
    m = wave_max(m);
    float ssum = 0.f;
    for (int p = p0 + lane; p < p1; p += 32) ssum += expf(elog[eid[p]] - m);
    ssum = wave_sum(ssum);
    float inv = 1.f / (ssum + 1e-16f);
    for (int p = p0 + lane; p < p1; p += 32) {
        int e = eid[p];
        elog[e] = expf(elog[e] - m) * inv;
    }
}

// ---------------------------------------------------------------------------
// Gather aggregation: one block per dst node, zero atomics.
// mode 0 = GCN (w = dinv[s]*dinv[d], self = dinv[d]^2)
// mode 1 = GAT (w = alpha[e],       self = aself[d])
// mode 2 = TR  (w = alpha[e],       no self term; out holds skip)
// ---------------------------------------------------------------------------
__global__ __launch_bounds__(256)
void node_agg_k(const int* __restrict__ row_start, const int* __restrict__ eid,
                const int* __restrict__ src, int D,
                const float* __restrict__ w_edge, const float* __restrict__ dinv,
                const float* __restrict__ wself, int mode,
                const float* __restrict__ H, float* __restrict__ out) {
    int d = blockIdx.x;
    int p0 = row_start[d], p1 = row_start[d + 1];
    int t = threadIdx.x;
    float dd = (mode == 0) ? dinv[d] : 0.f;
    for (int j = t; j < D; j += 256) {
        float acc = out[(size_t)d * D + j];             // bias / skip prefilled
        if (mode == 0)      acc += dd * dd * H[(size_t)d * D + j];
        else if (mode == 1) acc += wself[d] * H[(size_t)d * D + j];
        for (int p = p0; p < p1; ++p) {
            int e = eid[p];
            int s = src[e];
            float wgt = (mode == 0) ? dd * dinv[s] : w_edge[e];
            acc += wgt * H[(size_t)s * D + j];
        }
        out[(size_t)d * D + j] = acc;
    }
}

// ---------------------------------------------------------------------------
// Host orchestration
// ---------------------------------------------------------------------------
static inline unsigned gsz(long long n, int b) { return (unsigned)((n + b - 1) / b); }

extern "C" void kernel_launch(void* const* d_in, const int* in_sizes, int n_in,
                              void* d_out, int out_size, void* d_ws, size_t ws_size,
                              hipStream_t stream) {
    const int N = GN, IN = 1024, HID = 512;
    const float* x  = (const float*)d_in[0];
    const int*   ei = (const int*)d_in[1];
    const int    E  = in_sizes[1] / 2;
    const int* src = ei;
    const int* dst = ei + E;

    // params, flattened depth-first in dict order
    const float* gcn1W = (const float*)d_in[2];  const float* gcn1b = (const float*)d_in[3];
    const float* gcn2W = (const float*)d_in[4];  const float* gcn2b = (const float*)d_in[5];
    const float* t3qW  = (const float*)d_in[6];  const float* t3qb  = (const float*)d_in[7];
    const float* t3kW  = (const float*)d_in[8];  const float* t3kb  = (const float*)d_in[9];
    const float* t3vW  = (const float*)d_in[10]; const float* t3vb  = (const float*)d_in[11];
    const float* t3sW  = (const float*)d_in[12]; const float* t3sb  = (const float*)d_in[13];
    const float* t4qW  = (const float*)d_in[14]; const float* t4qb  = (const float*)d_in[15];
    const float* t4kW  = (const float*)d_in[16]; const float* t4kb  = (const float*)d_in[17];
    const float* t4vW  = (const float*)d_in[18]; const float* t4vb  = (const float*)d_in[19];
    const float* t4sW  = (const float*)d_in[20]; const float* t4sb  = (const float*)d_in[21];
    const float* g5W = (const float*)d_in[22]; const float* g5b = (const float*)d_in[23];
    const float* g5as = (const float*)d_in[24]; const float* g5ad = (const float*)d_in[25];
    const float* g6W = (const float*)d_in[26]; const float* g6b = (const float*)d_in[27];
    const float* g6as = (const float*)d_in[28]; const float* g6ad = (const float*)d_in[29];
    const float* g7W = (const float*)d_in[30]; const float* g7b = (const float*)d_in[31];
    const float* g7as = (const float*)d_in[32]; const float* g7ad = (const float*)d_in[33];
    const float* g8W = (const float*)d_in[34]; const float* g8b = (const float*)d_in[35];
    const float* g8as = (const float*)d_in[36]; const float* g8ad = (const float*)d_in[37];

    // output slices: h2, h4, h6, h8 each N x 1024
    float* out0 = (float*)d_out;
    float* out1 = out0 + (size_t)N * IN;
    float* out2 = out1 + (size_t)N * IN;
    float* out3 = out2 + (size_t)N * IN;

    // workspace carve-up
    const size_t NN = (size_t)N;
    float* qb   = (float*)d_ws;              // N x 1024 (also GEMM H for GCN/GAT)
    float* kb   = qb + NN * 1024;            // N x 1024
    float* vb   = kb + NN * 1024;            // N x 1024
    float* b512 = vb + NN * 1024;            // N x 512 (odd-layer activations)
    float* dinv = b512 + NN * 512;           // N
    float* asrc = dinv + NN;                 // N
    float* adst = asrc + NN;                 // N
    float* aslf = adst + NN;                 // N (GAT self alpha)
    float* alph = aslf + NN;                 // E (edge alpha / TR logits, in place)
    int* cnt       = (int*)(alph + (size_t)E);   // N  (counts, then cursor)
    int* row_start = cnt + NN;                   // N + 1
    int* eid       = row_start + NN + 1;         // E

    const int B = 256;

    auto gemm = [&](const float* A, const float* W, float* C, const float* bias,
                    int K, int Nc) {
        dim3 g(Nc / TN, N / TM), b(B);
        gemm_bf16_wmma<<<g, b, 0, stream>>>(A, W, C, bias, N, K, Nc);
    };

    // ---- CSR build + sym-norm degrees (once; shared by all layers) ----
    fill_i32<<<gsz(N, B), B, 0, stream>>>(cnt, N, 0);
    cnt_count_k<<<gsz(E, B), B, 0, stream>>>(dst, E, cnt);
    make_dinv_k<<<gsz(N, B), B, 0, stream>>>(cnt, dinv, N);
    scan4096_k<<<1, B, 0, stream>>>(cnt, row_start, N);
    fill_i32<<<gsz(N, B), B, 0, stream>>>(cnt, N, 0);        // reuse as cursor
    csr_scatter_k<<<gsz(E, B), B, 0, stream>>>(dst, E, row_start, cnt, eid);

    auto gcn_layer = [&](const float* X, const float* W, const float* bias,
                         float* out, int K, int Nc, bool act) {
        long long nd = (long long)N * Nc;
        gemm(X, W, qb, nullptr, K, Nc);
        fill_bias_k<<<gsz(nd, B), B, 0, stream>>>(out, bias, nd, Nc);
        node_agg_k<<<N, B, 0, stream>>>(row_start, eid, src, Nc,
                                        nullptr, dinv, nullptr, 0, qb, out);
        if (act) elu_k<<<gsz(nd, B), B, 0, stream>>>(out, nd);
    };

    auto gat_layer = [&](const float* X, const float* W, const float* bias,
                         const float* avs, const float* avd,
                         float* out, int K, int Nc, bool act) {
        long long nd = (long long)N * Nc;
        gemm(X, W, qb, nullptr, K, Nc);
        gat_alpha_k<<<gsz((long long)N * 32, B), B, 0, stream>>>(qb, avs, avd, N, Nc, asrc, adst);
        gat_softmax_k<<<gsz((long long)N * 32, B), B, 0, stream>>>(row_start, eid, src,
                                                                   asrc, adst, N, alph, aslf);
        fill_bias_k<<<gsz(nd, B), B, 0, stream>>>(out, bias, nd, Nc);
        node_agg_k<<<N, B, 0, stream>>>(row_start, eid, src, Nc,
                                        alph, nullptr, aslf, 1, qb, out);
        if (act) elu_k<<<gsz(nd, B), B, 0, stream>>>(out, nd);
    };

    auto tr_layer = [&](const float* X,
                        const float* Wq, const float* bq, const float* Wk, const float* bk,
                        const float* Wv, const float* bv, const float* Ws, const float* bs,
                        float* out, int K, int Nc, bool act) {
        long long nd = (long long)N * Nc;
        gemm(X, Wq, qb, bq, K, Nc);
        gemm(X, Wk, kb, bk, K, Nc);
        gemm(X, Wv, vb, bv, K, Nc);
        gemm(X, Ws, out, bs, K, Nc);   // skip connection pre-filled into out
        float scale = 1.0f / sqrtf((float)Nc);
        tr_logits_k<<<gsz((long long)E * 32, B), B, 0, stream>>>(src, dst, E, qb, kb, Nc, scale, alph);
        tr_softmax_k<<<gsz((long long)N * 32, B), B, 0, stream>>>(row_start, eid, N, alph);
        node_agg_k<<<N, B, 0, stream>>>(row_start, eid, src, Nc,
                                        alph, nullptr, nullptr, 2, vb, out);
        if (act) elu_k<<<gsz(nd, B), B, 0, stream>>>(out, nd);
    };

    // ---- 8-layer pipeline ----
    gcn_layer(x,    gcn1W, gcn1b, b512, IN,  HID, true);   // h1
    gcn_layer(b512, gcn2W, gcn2b, out0, HID, IN,  false);  // h2 -> out[0]
    tr_layer(out0, t3qW, t3qb, t3kW, t3kb, t3vW, t3vb, t3sW, t3sb, b512, IN,  HID, true);   // h3
    tr_layer(b512, t4qW, t4qb, t4kW, t4kb, t4vW, t4vb, t4sW, t4sb, out1, HID, IN,  false);  // h4 -> out[1]
    gat_layer(out1, g5W, g5b, g5as, g5ad, b512, IN,  HID, true);   // h5
    gat_layer(b512, g6W, g6b, g6as, g6ad, out2, HID, IN,  false);  // h6 -> out[2]
    gat_layer(out2, g7W, g7b, g7as, g7ad, b512, IN,  HID, true);   // h7
    gat_layer(b512, g8W, g8b, g8as, g8ad, out3, HID, IN,  false);  // h8 -> out[3]
}
